// GCN_9139690406275
// MI455X (gfx1250) — matrix operand order repaired and checked by reading, not compiled, not run on hardware
//
#include <hip/hip_runtime.h>
#include <hip/hip_bf16.h>
#include <cstdint>

#define NN 16000
#define DD 128

typedef __attribute__((ext_vector_type(16))) __bf16 v16bf;
typedef __attribute__((ext_vector_type(8)))  float  v8f;
typedef __attribute__((ext_vector_type(4)))  float  f32x4;   // clang vector for NT loads

union FragBF {
    uint4 q[2];
    v16bf v;
};

__device__ __forceinline__ uint16_t f2bf(float f) {
    uint32_t u = __float_as_uint(f);
    uint32_t r = ((u >> 16) & 1u) + 0x7FFFu;   // round-to-nearest-even
    return (uint16_t)((u + r) >> 16);
}

__device__ __forceinline__ void async_cp16(uint32_t lds_addr, uint64_t gaddr) {
    // async copy: 16B per lane, global -> LDS, tracked by ASYNCcnt
    asm volatile("global_load_async_to_lds_b128 %0, %1, off"
                 :: "v"(lds_addr), "v"(gaddr) : "memory");
}

__device__ __forceinline__ void wait_async0() {
    asm volatile("s_wait_asynccnt 0" ::: "memory");
}

// ---------------- h [N][128] f32 -> hT [128][N] bf16 ----------------
__global__ __launch_bounds__(256) void k_transpose_h(const float* __restrict__ h,
                                                     uint16_t* __restrict__ hbT) {
    __shared__ float tile[32][DD + 1];
    const int i0 = blockIdx.x * 32;
    const int t  = threadIdx.x;
#pragma unroll
    for (int s = 0; s < 16; ++s) {
        int idx = t + 256 * s;            // 0..4095
        int r = idx >> 7, c = idx & 127;
        tile[r][c] = h[(size_t)(i0 + r) * DD + c];
    }
    __syncthreads();
    const int kf = t >> 1, part = t & 1;  // 16 contiguous i per thread
    uint32_t o[8];
#pragma unroll
    for (int e = 0; e < 8; ++e) {
        float a = tile[part * 16 + 2 * e + 0][kf];
        float b = tile[part * 16 + 2 * e + 1][kf];
        o[e] = (uint32_t)f2bf(a) | ((uint32_t)f2bf(b) << 16);
    }
    uint4* dst = (uint4*)(hbT + (size_t)kf * NN + i0 + part * 16);
    dst[0] = make_uint4(o[0], o[1], o[2], o[3]);
    dst[1] = make_uint4(o[4], o[5], o[6], o[7]);
}

// ---------------- W [in][out] f32 -> Wt [out][in] bf16 ----------------
__global__ __launch_bounds__(256) void k_transpose_w(const float* __restrict__ W1,
                                                     const float* __restrict__ W2,
                                                     uint16_t* __restrict__ w1t,
                                                     uint16_t* __restrict__ w2t) {
    const float* W   = blockIdx.x ? W2 : W1;
    uint16_t*    Wt  = blockIdx.x ? w2t : w1t;
    for (int e = threadIdx.x; e < DD * DD; e += 256) {
        int in_ = e >> 7, out_ = e & 127;
        Wt[out_ * DD + in_] = f2bf(W[e]);
    }
}

__global__ __launch_bounds__(256) void k_zero(float4* __restrict__ p) {
    size_t tid = (size_t)blockIdx.x * 256 + threadIdx.x;
    p[tid] = make_float4(0.f, 0.f, 0.f, 0.f);
}

// ---------------- main: agg += maskT(g) @ h  (bf16 WMMA, f32 accum) ----------------
// Double-buffered LDS: buf k at k*20480: maskT 128x80B @0, hT-tile 128x80B @10240
__global__ __launch_bounds__(256) void k_agg(const float* __restrict__ g,
                                             const uint16_t* __restrict__ hbT,
                                             float* __restrict__ aggF) {
    extern __shared__ char smem[];

    const int jb   = blockIdx.x * 128;
    const int iBeg = blockIdx.y * (NN / 4);
    const int iEnd = iBeg + (NN / 4);
    const int t    = threadIdx.x;
    const int lane = t & 31, w = t >> 5;
    const int strip   = w * 16;
    const int halfSel = lane >> 4;     // 0/1
    const int lr      = lane & 15;
    const int kbA     = halfSel * 8;   // A-frag K base
    const int kbB     = halfSel * 16;  // unused; B frags also use kbA
    const uint64_t hbT_base = (uint64_t)hbT;

    auto stage_ht = [&](int i0, char* htB) {
#pragma unroll
        for (int s = 0; s < 2; ++s) {
            int c = t + 256 * s;                  // 0..511 chunks of 16B
            int kf = c >> 2, part = c & 3;
            uint32_t lds = (uint32_t)(uintptr_t)(htB + kf * 80 + part * 16);
            uint64_t ga  = hbT_base + (((uint64_t)kf * NN + (uint64_t)i0) + (uint64_t)part * 8) * 2;
            async_cp16(lds, ga);
        }
    };
    auto load_g = [&](int i0, f32x4* gv) {
#pragma unroll
        for (int s = 0; s < 4; ++s) {
            int q = t + 256 * s;                  // 0..1023 float4s
            int r = q >> 5;                       // i offset 0..31
            int joff = (q & 31) * 4;
            gv[s] = __builtin_nontemporal_load(
                (const f32x4*)(g + (size_t)(i0 + r) * NN + jb + joff));
        }
    };
    auto store_mask = [&](char* maskB, const f32x4* gv) {
#pragma unroll
        for (int s = 0; s < 4; ++s) {
            int q = t + 256 * s;
            int r = q >> 5;
            int joff = (q & 31) * 4;
            *(uint16_t*)(maskB + (size_t)(joff + 0) * 80 + r * 2) = (gv[s].x > 0.f) ? 0x3F80 : 0;
            *(uint16_t*)(maskB + (size_t)(joff + 1) * 80 + r * 2) = (gv[s].y > 0.f) ? 0x3F80 : 0;
            *(uint16_t*)(maskB + (size_t)(joff + 2) * 80 + r * 2) = (gv[s].z > 0.f) ? 0x3F80 : 0;
            *(uint16_t*)(maskB + (size_t)(joff + 3) * 80 + r * 2) = (gv[s].w > 0.f) ? 0x3F80 : 0;
        }
    };

    v8f acc[8];
#pragma unroll
    for (int n = 0; n < 8; ++n) acc[n] = (v8f){0.f, 0.f, 0.f, 0.f, 0.f, 0.f, 0.f, 0.f};

    // ---- prologue: stage tile 0 into buffer 0 ----
    f32x4 gv[4];
    stage_ht(iBeg, smem + 10240);
    load_g(iBeg, gv);
    store_mask(smem, gv);
    wait_async0();
    __syncthreads();

    int cur = 0;
    for (int i0 = iBeg; i0 < iEnd; i0 += 32) {
        char* maskB = smem + cur * 20480;
        char* htB   = maskB + 10240;
        char* maskN = smem + (cur ^ 1) * 20480;
        char* htN   = maskN + 10240;
        const bool hasNext = (i0 + 32) < iEnd;

        // ---- issue next tile's memory ops before computing (hide latency) ----
        if (hasNext) {
            stage_ht(i0 + 32, htN);            // async, ASYNCcnt
            load_g(i0 + 32, gv);               // NT global loads into regs
            if (i0 + 64 < iEnd) {              // prefetch tile after next
                const char* p = (const char*)(g + (size_t)(i0 + 64 + (t >> 3)) * NN +
                                              jb + (t & 7) * 16);
                __builtin_prefetch(p, 0, 0);
            }
        }

        // ---- compute on current buffers: 8 WMMAs, B-frags preloaded 4 at a time ----
        FragBF a;
        const char* arow = maskB + (size_t)(strip + lr) * 80 + kbA * 2;
        a.q[0] = *(const uint4*)arow;
        a.q[1] = *(const uint4*)(arow + 32);
#pragma unroll
        for (int half = 0; half < 2; ++half) {
            FragBF b[4];
#pragma unroll
            for (int n = 0; n < 4; ++n) {
                const char* brow = htB + (size_t)((half * 4 + n) * 16 + lr) * 80 + kbA * 2;
                b[n].q[0] = *(const uint4*)brow;
                b[n].q[1] = *(const uint4*)(brow + 32);
            }
#pragma unroll
            for (int n = 0; n < 4; ++n) {
                acc[half * 4 + n] = __builtin_amdgcn_wmma_f32_16x16x32_bf16(
                    false, a.v, false, b[n].v, (short)0, acc[half * 4 + n], false, false);
            }
        }

        // ---- finish next tile's staging, then single barrier ----
        if (hasNext) store_mask(maskN, gv);
        wait_async0();
        __syncthreads();
        cur ^= 1;
    }

    // ---- merge partial sums (4 i-segments) ----
#pragma unroll
    for (int n = 0; n < 8; ++n) {
#pragma unroll
        for (int r = 0; r < 8; ++r) {
            int m   = r + halfSel * 8;
            int row = jb + strip + m;
            int col = n * 16 + lr;
            atomicAdd(&aggF[(size_t)row * DD + col], acc[n][r]);
        }
    }
}

// ---------------- y0 = bf16(h + agg) ----------------
__global__ __launch_bounds__(256) void k_y0(const float* __restrict__ h,
                                            const float* __restrict__ agg,
                                            uint16_t* __restrict__ y0) {
    size_t tid = (size_t)blockIdx.x * 256 + threadIdx.x;
    size_t i4  = tid * 4;
    float4 hv = *(const float4*)(h + i4);
    float4 av = *(const float4*)(agg + i4);
    uint32_t p0 = (uint32_t)f2bf(hv.x + av.x) | ((uint32_t)f2bf(hv.y + av.y) << 16);
    uint32_t p1 = (uint32_t)f2bf(hv.z + av.z) | ((uint32_t)f2bf(hv.w + av.w) << 16);
    *(uint2*)(y0 + i4) = make_uint2(p0, p1);
}

// ---------------- MLP: relu(relu(y@W1+b1)@W2+b2), WMMA bf16 ----------------
// LDS: yBuf 128 rows x 272B (128 bf16 + 16B pad)
__global__ __launch_bounds__(256) void k_mlp(const uint16_t* __restrict__ y0,
                                             const uint16_t* __restrict__ w1t,
                                             const uint16_t* __restrict__ w2t,
                                             const float* __restrict__ b1,
                                             const float* __restrict__ b2,
                                             float* __restrict__ out) {
    extern __shared__ char smem[];
    const int rb   = blockIdx.x * 128;
    const int t    = threadIdx.x;
    const int lane = t & 31, w = t >> 5;
    const int strip   = w * 16;
    const int halfSel = lane >> 4;
    const int lr      = lane & 15;
    const int kbA     = halfSel * 8;
    const int kbB     = halfSel * 16;

    // async-stage this block's 128 y-rows (bf16, row stride 256B -> padded 272B)
    const uint64_t ybase = (uint64_t)(y0 + (size_t)rb * DD);
#pragma unroll
    for (int s = 0; s < 8; ++s) {
        int c = t + 256 * s;                 // 0..2047 chunks of 16B
        int row = c >> 4, part = c & 15;
        uint32_t lds = (uint32_t)(uintptr_t)(smem + row * 272 + part * 16);
        async_cp16(lds, ybase + (uint64_t)row * 256 + (uint64_t)part * 16);
    }
    wait_async0();
    __syncthreads();

    float b1v[8], b2v[8];
#pragma unroll
    for (int n = 0; n < 8; ++n) { b1v[n] = b1[n * 16 + lr]; b2v[n] = b2[n * 16 + lr]; }

    // ---- GEMM1: t1 = y @ W1 ----
    v8f acc[8];
#pragma unroll
    for (int n = 0; n < 8; ++n) acc[n] = (v8f){0.f, 0.f, 0.f, 0.f, 0.f, 0.f, 0.f, 0.f};
#pragma unroll
    for (int kc = 0; kc < DD; kc += 32) {
        FragBF a;
        const char* arow = smem + (size_t)(strip + lr) * 272 + (kc + kbA) * 2;
        a.q[0] = *(const uint4*)arow;
        a.q[1] = *(const uint4*)(arow + 32);
#pragma unroll
        for (int n = 0; n < 8; ++n) {
            FragBF b;
            const uint16_t* brow = w1t + (size_t)(n * 16 + lr) * DD + kc + kbA;
            b.q[0] = *(const uint4*)brow;
            b.q[1] = *(const uint4*)(brow + 16);
            acc[n] = __builtin_amdgcn_wmma_f32_16x16x32_bf16(false, a.v, false, b.v,
                                                             (short)0, acc[n], false, false);
        }
    }
    // bias + relu, write back into this wave's own LDS rows (per-wave DS ordering)
#pragma unroll
    for (int n = 0; n < 8; ++n) {
#pragma unroll
        for (int r = 0; r < 8; ++r) {
            int m = r + halfSel * 8;
            float v = acc[n][r] + b1v[n];
            v = v > 0.f ? v : 0.f;
            *(uint16_t*)(smem + (size_t)(strip + m) * 272 + (n * 16 + lr) * 2) = f2bf(v);
        }
    }

    // ---- GEMM2: y2 = t1 @ W2 ----
    v8f acc2[8];
#pragma unroll
    for (int n = 0; n < 8; ++n) acc2[n] = (v8f){0.f, 0.f, 0.f, 0.f, 0.f, 0.f, 0.f, 0.f};
#pragma unroll
    for (int kc = 0; kc < DD; kc += 32) {
        FragBF a;
        const char* arow = smem + (size_t)(strip + lr) * 272 + (kc + kbA) * 2;
        a.q[0] = *(const uint4*)arow;
        a.q[1] = *(const uint4*)(arow + 32);
#pragma unroll
        for (int n = 0; n < 8; ++n) {
            FragBF b;
            const uint16_t* brow = w2t + (size_t)(n * 16 + lr) * DD + kc + kbA;
            b.q[0] = *(const uint4*)brow;
            b.q[1] = *(const uint4*)(brow + 16);
            acc2[n] = __builtin_amdgcn_wmma_f32_16x16x32_bf16(false, a.v, false, b.v,
                                                              (short)0, acc2[n], false, false);
        }
    }
    // bias + relu -> f32 output
#pragma unroll
    for (int n = 0; n < 8; ++n) {
#pragma unroll
        for (int r = 0; r < 8; ++r) {
            int m = r + halfSel * 8;
            float v = acc2[n][r] + b2v[n];
            v = v > 0.f ? v : 0.f;
            out[(size_t)(rb + strip + m) * DD + n * 16 + lr] = v;
        }
    }
}

extern "C" void kernel_launch(void* const* d_in, const int* in_sizes, int n_in,
                              void* d_out, int out_size, void* d_ws, size_t ws_size,
                              hipStream_t stream) {
    const float* g  = (const float*)d_in[0];
    const float* h  = (const float*)d_in[1];
    const float* W1 = (const float*)d_in[2];
    const float* b1 = (const float*)d_in[3];
    const float* W2 = (const float*)d_in[4];
    const float* b2 = (const float*)d_in[5];
    float* out = (float*)d_out;

    char* ws = (char*)d_ws;
    uint16_t* hbT = (uint16_t*)(ws);               // [128][16000] bf16 : 4,096,000 B
    uint16_t* y0  = (uint16_t*)(ws + 4096000);     // [16000][128] bf16 : 4,096,000 B
    float*    agg = (float*)   (ws + 8192000);     // [16000][128] f32  : 8,192,000 B
    uint16_t* w1t = (uint16_t*)(ws + 16384000);    // [128][128] bf16   : 32,768 B
    uint16_t* w2t = (uint16_t*)(ws + 16416768);    // [128][128] bf16   : 32,768 B

    k_transpose_h<<<NN / 32, 256, 0, stream>>>(h, hbT);
    k_transpose_w<<<2, 256, 0, stream>>>(W1, W2, w1t, w2t);
    k_zero<<<2000, 256, 0, stream>>>((float4*)agg);                   // (NN*DD)/4 float4
    k_agg<<<dim3(NN / 128, 4), 256, 40960, stream>>>(g, hbT, agg);    // double-buffered LDS
    k_y0<<<2000, 256, 0, stream>>>(h, agg, y0);
    k_mlp<<<NN / 128, 256, 34816, stream>>>(y0, w1t, w2t, b1, b2, out);
}
